// MultiHeadSelfAttention_23708219474166
// MI455X (gfx1250) — matrix operand order repaired and checked by reading, not compiled, not run on hardware
//
#include <hip/hip_runtime.h>
#include <hip/hip_bf16.h>
#include <cstdint>
#include <cstddef>

// ---------------------------------------------------------------------------
// MHSA for gfx1250 (MI455X): B=4, T=1024, D_MODEL=512, N_HEADS=8, head_dim=512
//   y = softmax( (X Wq^T)_h (X Wk^T)_h^T * 512^-0.5 ) (X Wv^T)_h
// All GEMMs run through v_wmma_f32_16x16x32_bf16 (f32 accumulate).
// Wave tile 32x64 (2x4 WMMA accum tiles): 8 WMMAs per 6 fragment loads.
// Workspace layout (bytes), total 240 MiB (assumed available):
//   [0,4M)    x   bf16  [4096,512]
//   [4M,8M)   Wq  bf16  [4096,512]
//   [8M,12M)  Wk  bf16  [4096,512]
//   [12M,16M) Wv  bf16  [4096,512]
//   [16M,48M)  Q  bf16  [b,h,t,d]   (pre-scaled by 512^-0.5)
//   [48M,80M)  K  bf16  [b,h,t,d]
//   [80M,112M) V^T bf16 [b,h,d,t]
//   [112M,240M) S f32   [b,h,tq,tk]; softmax overwrites each row in-place
//               with bf16 P (row stride 2048 ushorts, first 1024 valid)
// ---------------------------------------------------------------------------

typedef __attribute__((ext_vector_type(16))) __bf16 v16bf;
typedef __attribute__((ext_vector_type(8)))  float  v8f;

#define HEADS   8
#define DMODEL  512
#define BATCH   4
#define TLEN    1024
#define BT      (BATCH * TLEN)     // 4096
#define PROJ    (DMODEL * HEADS)   // 4096

__device__ __forceinline__ unsigned short f2bf(float f) {
    unsigned u = __float_as_uint(f);
    u += 0x7FFFu + ((u >> 16) & 1u);          // round-to-nearest-even
    return (unsigned short)(u >> 16);
}

union FragCvt { uint4 u[2]; v16bf v; };

// A-fragment, 16x32 bf16 (M x K): lane = M%16; per-lane halves:
//   j in [0,15] -> K = k0 + (lane<16?0:16) + j   (one contiguous 32B span)
__device__ __forceinline__ v16bf load_frag_a(const unsigned short* __restrict__ base,
                                             int ld, int row0, int k0) {
    const int lane = threadIdx.x & 31;
    const unsigned short* p =
        base + (size_t)(row0 + (lane & 15)) * ld + k0 + (lane & 16);
    FragCvt f;
    f.u[0] = *(const uint4*)(p);
    f.u[1] = *(const uint4*)(p + 8);
    return f.v;
}

// B-fragment, 32x16 bf16 (K x N): lane = N%16; per-lane halves:
//   j in [0,15] -> K = k0 + (lane<16?0:16) + j   (one contiguous 32B span)
__device__ __forceinline__ v16bf load_frag_b(const unsigned short* __restrict__ base,
                                             int ld, int col0, int k0) {
    const int lane = threadIdx.x & 31;
    const unsigned short* p =
        base + (size_t)(col0 + (lane & 15)) * ld + k0 + (lane & 16);
    FragCvt f;
    f.u[0] = *(const uint4*)(p);
    f.u[1] = *(const uint4*)(p + 8);
    return f.v;
}

__device__ __forceinline__ v8f wmma_bf16(v16bf a, v16bf b, v8f c) {
    return __builtin_amdgcn_wmma_f32_16x16x32_bf16(
        /*neg_a=*/false, a, /*neg_b=*/false, b,
        /*c_mod=*/(short)0, c, /*reuse_a=*/false, /*reuse_b=*/false);
}

// One wave computes a 32x64 f32 tile: rows m0..m0+31 of A (row-major [m,k]),
// cols n0..n0+63 of B (row-major [n,k], i.e. B^T of the math GEMM).
// 8 WMMAs per k-step, 6 fragment loads.
__device__ __forceinline__ void wave_gemm_32x64(
    const unsigned short* __restrict__ A, int lda, int m0,
    const unsigned short* __restrict__ B, int ldb, int n0,
    int K, v8f (&c)[2][4])
{
    const int lane = threadIdx.x & 31;
    for (int k0 = 0; k0 < K; k0 += 32) {
        if (k0 + 64 <= K) {   // pull next-next tile toward L2/WGP$
            __builtin_prefetch(A + (size_t)(m0 + lane) * lda + k0 + 64, 0, 1);
            __builtin_prefetch(B + (size_t)(n0 + lane) * ldb + k0 + 64, 0, 1);
            __builtin_prefetch(B + (size_t)(n0 + 32 + lane) * ldb + k0 + 64, 0, 1);
        }
        v16bf a0 = load_frag_a(A, lda, m0,      k0);
        v16bf a1 = load_frag_a(A, lda, m0 + 16, k0);
        v16bf b0 = load_frag_b(B, ldb, n0,      k0);
        v16bf b1 = load_frag_b(B, ldb, n0 + 16, k0);
        v16bf b2 = load_frag_b(B, ldb, n0 + 32, k0);
        v16bf b3 = load_frag_b(B, ldb, n0 + 48, k0);
        c[0][0] = wmma_bf16(a0, b0, c[0][0]);
        c[0][1] = wmma_bf16(a0, b1, c[0][1]);
        c[0][2] = wmma_bf16(a0, b2, c[0][2]);
        c[0][3] = wmma_bf16(a0, b3, c[0][3]);
        c[1][0] = wmma_bf16(a1, b0, c[1][0]);
        c[1][1] = wmma_bf16(a1, b1, c[1][1]);
        c[1][2] = wmma_bf16(a1, b2, c[1][2]);
        c[1][3] = wmma_bf16(a1, b3, c[1][3]);
    }
}

// ---------------------------------------------------------------------------
// Kernel 1: fp32 -> bf16 conversion for x, Wq, Wk, Wv (grid.y selects array)
// ---------------------------------------------------------------------------
__global__ __launch_bounds__(256)
void cvt_f32_bf16_kernel(const float* __restrict__ s0, const float* __restrict__ s1,
                         const float* __restrict__ s2, const float* __restrict__ s3,
                         unsigned short* __restrict__ d0, unsigned short* __restrict__ d1,
                         unsigned short* __restrict__ d2, unsigned short* __restrict__ d3,
                         int n)
{
    const float* s;
    unsigned short* d;
    switch (blockIdx.y) {
        case 0:  s = s0; d = d0; break;
        case 1:  s = s1; d = d1; break;
        case 2:  s = s2; d = d2; break;
        default: s = s3; d = d3; break;
    }
    int i = (int)(blockIdx.x * blockDim.x + threadIdx.x) * 4;
    if (i >= n) return;
    float4 v = *(const float4*)(s + i);
    ushort4 o;
    o.x = f2bf(v.x); o.y = f2bf(v.y); o.z = f2bf(v.z); o.w = f2bf(v.w);
    *(ushort4*)(d + i) = o;
}

// ---------------------------------------------------------------------------
// Kernel 2: projection GEMM Y = X @ W^T, M=N=4096, K=512.
// grid.z: 0 -> Q (scaled, [b,h,t,d]), 1 -> K ([b,h,t,d]), 2 -> V^T ([b,h,d,t])
// split_heads: head h = n & 7, d = n >> 3 (h is the fast factor).
// ---------------------------------------------------------------------------
__global__ __launch_bounds__(256)
void proj_kernel(const unsigned short* __restrict__ xb,
                 const unsigned short* __restrict__ wqb,
                 const unsigned short* __restrict__ wkb,
                 const unsigned short* __restrict__ wvb,
                 unsigned short* __restrict__ Qo,
                 unsigned short* __restrict__ Ko,
                 unsigned short* __restrict__ VTo)
{
    const int mode = blockIdx.z;
    const unsigned short* W = (mode == 0) ? wqb : (mode == 1) ? wkb : wvb;
    const int wid = threadIdx.x >> 5;
    const int m0 = blockIdx.x * 128 + (wid & 3) * 32;
    const int n0 = blockIdx.y * 128 + (wid >> 2) * 64;

    v8f c[2][4] = {};
    wave_gemm_32x64(xb, DMODEL, m0, W, DMODEL, n0, DMODEL, c);

    const int lane = threadIdx.x & 31;
    const int rsel = (lane & 16) >> 1;   // +0 / +8 row
    const int csel = lane & 15;
    const float scale = (mode == 0) ? 0.044194173824159216f : 1.0f; // 512^-0.5

    #pragma unroll
    for (int rt = 0; rt < 2; ++rt) {
        const int mbase = m0 + rt * 16 + rsel;
        #pragma unroll
        for (int ct = 0; ct < 4; ++ct) {
            const int n = n0 + ct * 16 + csel;
            const int h = n & 7, dd = n >> 3;
            #pragma unroll
            for (int r = 0; r < 8; ++r) {
                const int m  = mbase + r;
                const int bb = m >> 10, tt = m & 1023;
                const size_t bh = (size_t)(bb * HEADS + h);
                const unsigned short bv = f2bf(c[rt][ct][r] * scale);
                if (mode == 0)      Qo [(bh * TLEN + tt) * DMODEL + dd] = bv;
                else if (mode == 1) Ko [(bh * TLEN + tt) * DMODEL + dd] = bv;
                else                VTo[(bh * DMODEL + dd) * TLEN + tt] = bv;
            }
        }
    }
}

// ---------------------------------------------------------------------------
// Kernel 3: scores S = Q K^T per (b,h). M=N=1024, K=512. f32 output.
// ---------------------------------------------------------------------------
__global__ __launch_bounds__(256)
void scores_kernel(const unsigned short* __restrict__ Q,
                   const unsigned short* __restrict__ Kt,
                   float* __restrict__ S)
{
    const int bh = blockIdx.z;
    const unsigned short* Qb = Q  + (size_t)bh * TLEN * DMODEL;
    const unsigned short* Kb = Kt + (size_t)bh * TLEN * DMODEL;
    float* Sb = S + (size_t)bh * TLEN * TLEN;

    const int wid = threadIdx.x >> 5;
    const int m0 = blockIdx.x * 128 + (wid & 3) * 32;
    const int n0 = blockIdx.y * 128 + (wid >> 2) * 64;

    v8f c[2][4] = {};
    wave_gemm_32x64(Qb, DMODEL, m0, Kb, DMODEL, n0, DMODEL, c);

    const int lane = threadIdx.x & 31;
    const int rsel = (lane & 16) >> 1;
    const int csel = lane & 15;
    #pragma unroll
    for (int rt = 0; rt < 2; ++rt) {
        const int m = m0 + rt * 16 + rsel;
        #pragma unroll
        for (int ct = 0; ct < 4; ++ct) {
            const int n = n0 + ct * 16 + csel;
            #pragma unroll
            for (int r = 0; r < 8; ++r)
                Sb[(size_t)(m + r) * TLEN + n] = c[rt][ct][r];
        }
    }
}

// ---------------------------------------------------------------------------
// Kernel 4: row softmax. One wave per 1024-wide row; overwrite the row start
// in-place with bf16 P (row stride stays 4096 bytes = 2048 ushorts).
// ---------------------------------------------------------------------------
__global__ __launch_bounds__(256)
void softmax_kernel(float* __restrict__ S)
{
    const int row  = (int)blockIdx.x * 8 + (int)(threadIdx.x >> 5);
    const int lane = threadIdx.x & 31;
    float* rp = S + (size_t)row * TLEN;

    float v[32];
    float m = -3.402823466e38f;
    #pragma unroll
    for (int j = 0; j < 32; ++j) { v[j] = rp[lane + 32 * j]; m = fmaxf(m, v[j]); }
    #pragma unroll
    for (int o = 16; o > 0; o >>= 1) m = fmaxf(m, __shfl_xor(m, o, 32));

    float s = 0.f;
    #pragma unroll
    for (int j = 0; j < 32; ++j) { v[j] = __expf(v[j] - m); s += v[j]; }
    #pragma unroll
    for (int o = 16; o > 0; o >>= 1) s += __shfl_xor(s, o, 32);

    const float inv = 1.0f / s;
    unsigned short* pp = (unsigned short*)rp;   // bf16 P overlays the f32 row
    #pragma unroll
    for (int j = 0; j < 32; ++j) pp[lane + 32 * j] = f2bf(v[j] * inv);
}

// ---------------------------------------------------------------------------
// Kernel 5: O = P @ V per (b,h). M=1024, N=512, K=1024.
// A = P (bf16 overlay, lda = 2048 ushorts), B rows from V^T [d, t].
// out[b, t, h*512 + d], fp32.
// ---------------------------------------------------------------------------
__global__ __launch_bounds__(256)
void pv_kernel(const unsigned short* __restrict__ P,
               const unsigned short* __restrict__ VT,
               float* __restrict__ out)
{
    const int bh = blockIdx.z;
    const int b = bh >> 3, h = bh & 7;
    const unsigned short* Pb = P  + (size_t)bh * TLEN * 2048;
    const unsigned short* Vb = VT + (size_t)bh * DMODEL * TLEN;

    const int wid = threadIdx.x >> 5;
    const int m0 = blockIdx.x * 128 + (wid & 3) * 32;
    const int n0 = blockIdx.y * 128 + (wid >> 2) * 64;

    v8f c[2][4] = {};
    wave_gemm_32x64(Pb, 2048, m0, Vb, TLEN, n0, TLEN, c);

    const int lane = threadIdx.x & 31;
    const int rsel = (lane & 16) >> 1;
    const int csel = lane & 15;
    #pragma unroll
    for (int rt = 0; rt < 2; ++rt) {
        const int m = m0 + rt * 16 + rsel;
        #pragma unroll
        for (int ct = 0; ct < 4; ++ct) {
            const int n = n0 + ct * 16 + csel;
            #pragma unroll
            for (int r = 0; r < 8; ++r) {
                const int tq = m + r;
                out[(size_t)(b * TLEN + tq) * (HEADS * DMODEL) + h * DMODEL + n] = c[rt][ct][r];
            }
        }
    }
}

// ---------------------------------------------------------------------------
extern "C" void kernel_launch(void* const* d_in, const int* in_sizes, int n_in,
                              void* d_out, int out_size, void* d_ws, size_t ws_size,
                              hipStream_t stream)
{
    // setup_inputs order: x, Wk, Wv, Wq
    const float* x  = (const float*)d_in[0];
    const float* Wk = (const float*)d_in[1];
    const float* Wv = (const float*)d_in[2];
    const float* Wq = (const float*)d_in[3];
    float* out = (float*)d_out;

    char* ws = (char*)d_ws;
    const size_t MB = 1u << 20;
    unsigned short* xb  = (unsigned short*)(ws + 0 * MB);
    unsigned short* wqb = (unsigned short*)(ws + 4 * MB);
    unsigned short* wkb = (unsigned short*)(ws + 8 * MB);
    unsigned short* wvb = (unsigned short*)(ws + 12 * MB);
    unsigned short* Qb  = (unsigned short*)(ws + 16 * MB);
    unsigned short* Kb  = (unsigned short*)(ws + 48 * MB);
    unsigned short* VTb = (unsigned short*)(ws + 80 * MB);
    float*          Sb  = (float*)        (ws + 112 * MB);   // 128 MiB

    const int nconv = BT * DMODEL;   // 2,097,152 elements per array
    cvt_f32_bf16_kernel<<<dim3(nconv / (256 * 4), 4), 256, 0, stream>>>(
        x, Wq, Wk, Wv, xb, wqb, wkb, wvb, nconv);

    proj_kernel<<<dim3(BT / 128, PROJ / 128, 3), 256, 0, stream>>>(
        xb, wqb, wkb, wvb, Qb, Kb, VTb);

    scores_kernel<<<dim3(TLEN / 128, TLEN / 128, BATCH * HEADS), 256, 0, stream>>>(
        Qb, Kb, Sb);

    softmax_kernel<<<dim3(BATCH * HEADS * TLEN / 8), 256, 0, stream>>>(Sb);

    pv_kernel<<<dim3(TLEN / 128, DMODEL / 128, BATCH * HEADS), 256, 0, stream>>>(
        (const unsigned short*)Sb, VTb, out);
}
